// Attention_9998683865539
// MI455X (gfx1250) — matrix operand order repaired and checked by reading, not compiled, not run on hardware
//
#include <hip/hip_runtime.h>

#define BB 8
#define NN 1024
#define CC 768
#define HH 12
#define HD 64
#define TOKENS (BB * NN)   // 8192
#define C3 (3 * CC)        // 2304

typedef __attribute__((ext_vector_type(4)))  __bf16 v4bf;
typedef __attribute__((ext_vector_type(8)))  __bf16 v8bf;
typedef __attribute__((ext_vector_type(16))) __bf16 v16bf;
typedef __attribute__((ext_vector_type(4)))  float  v4f;
typedef __attribute__((ext_vector_type(8)))  float  v8f;

// ---------------------------------------------------------------------------
// WMMA fragment loaders (bf16, 16x16x32).
// A (16xK rows m, k): lane L (r=L&15, h=L>>4) holds row m0+r,
//   k = {k0+h*8 .. +7} in elems 0..7 and {k0+h*8+16 .. +7} in elems 8..15.
// B (Kx16, cols n): lane L holds column n0+r, same k order as A
//   ({k0+h*8 .. +7} in elems 0..7, {k0+h*8+16 .. +7} in elems 8..15).
//   srcT is the row-major [Ncols x K] view (i.e. B transposed).
// ---------------------------------------------------------------------------
static __device__ inline v16bf load_a_frag(const __bf16* __restrict__ src,
                                           int ld, int m0, int k0, int lane) {
  const int r = lane & 15, h = lane >> 4;
  const __bf16* base = src + (m0 + r) * ld + k0 + h * 8;
  v8bf lo = *(const v8bf*)(base);
  v8bf hi = *(const v8bf*)(base + 16);
  v16bf out;
#pragma unroll
  for (int i = 0; i < 8; ++i) { out[i] = lo[i]; out[i + 8] = hi[i]; }
  return out;
}

static __device__ inline v16bf load_b_frag(const __bf16* __restrict__ srcT,
                                           int ld, int n0, int k0, int lane) {
  const int r = lane & 15, h = lane >> 4;
  const __bf16* base = srcT + (n0 + r) * ld + k0 + h * 8;
  v8bf lo = *(const v8bf*)(base);
  v8bf hi = *(const v8bf*)(base + 16);
  v16bf out;
#pragma unroll
  for (int i = 0; i < 8; ++i) { out[i] = lo[i]; out[i + 8] = hi[i]; }
  return out;
}

static __device__ inline v8f wmma_bf16(v16bf a, v16bf b, v8f c) {
  return __builtin_amdgcn_wmma_f32_16x16x32_bf16(false, a, false, b,
                                                 (short)0, c, false, false);
}

// ---------------------------------------------------------------------------
// f32 -> bf16 conversion (vector x4)
// ---------------------------------------------------------------------------
__global__ __launch_bounds__(256) void cvt_f32_bf16(const float* __restrict__ in,
                                                    __bf16* __restrict__ out,
                                                    int n4) {
  int i = blockIdx.x * 256 + threadIdx.x;
  if (i >= n4) return;
  v4f f = ((const v4f*)in)[i];
  v4bf o;
#pragma unroll
  for (int j = 0; j < 4; ++j) o[j] = (__bf16)f[j];
  ((v4bf*)out)[i] = o;
}

// ---------------------------------------------------------------------------
// QKV GEMM: [8192 x 768] x [768 x 2304] (B given as w_qkv[n][k] row-major).
// Block = 256 thr = 8 waves (4 M x 2 N), block tile 128x128, wave tile 32x64.
// Epilogue scatters into q [B,H,N,64] (pre-scaled by 1/8), k [B,H,N,64],
// and V transposed vT [B,H,64,N].
// ---------------------------------------------------------------------------
__global__ __launch_bounds__(256) void qkv_gemm(const __bf16* __restrict__ xb,
                                                const __bf16* __restrict__ wq,
                                                __bf16* __restrict__ qg,
                                                __bf16* __restrict__ kg,
                                                __bf16* __restrict__ vtg) {
  const int lane = threadIdx.x & 31;
  const int wave = threadIdx.x >> 5;
  const int wm = wave & 3, wn = wave >> 2;
  const int m0 = blockIdx.x * 128 + wm * 32;
  const int n0 = blockIdx.y * 128 + wn * 64;

  v8f acc[2][4] = {};
  for (int k0 = 0; k0 < CC; k0 += 32) {
    const v16bf a0 = load_a_frag(xb, CC, m0, k0, lane);
    const v16bf a1 = load_a_frag(xb, CC, m0 + 16, k0, lane);
#pragma unroll
    for (int ni = 0; ni < 4; ++ni) {
      const v16bf b = load_b_frag(wq, CC, n0 + ni * 16, k0, lane);
      acc[0][ni] = wmma_bf16(a0, b, acc[0][ni]);
      acc[1][ni] = wmma_bf16(a1, b, acc[1][ni]);
    }
  }

  const int r = lane & 15, h = lane >> 4;
#pragma unroll
  for (int mi = 0; mi < 2; ++mi) {
#pragma unroll
    for (int ni = 0; ni < 4; ++ni) {
#pragma unroll
      for (int g = 0; g < 8; ++g) {
        const int M = m0 + mi * 16 + g + 8 * h;      // token index
        const int Ncol = n0 + ni * 16 + r;           // 0..2303
        const float val = acc[mi][ni][g];
        const int which = Ncol / CC;
        const int c = Ncol % CC;
        const int head = c >> 6, d = c & 63;
        const int b_ = M >> 10, tok = M & 1023;
        if (which == 0) {
          qg[((b_ * HH + head) * NN + tok) * HD + d] = (__bf16)(val * 0.125f);
        } else if (which == 1) {
          kg[((b_ * HH + head) * NN + tok) * HD + d] = (__bf16)val;
        } else {
          vtg[((b_ * HH + head) * HD + d) * NN + tok] = (__bf16)val;
        }
      }
    }
  }
}

// ---------------------------------------------------------------------------
// Fused flash attention. Grid = B*H*(N/128) blocks of 256 threads.
// Each wave owns 16 query rows; streams keys in blocks of 32 with online
// softmax; P goes through a 1KB/wave LDS tile to re-layout C/D -> A fragment.
// ---------------------------------------------------------------------------
__global__ __launch_bounds__(256) void attn_fused(const __bf16* __restrict__ qg,
                                                  const __bf16* __restrict__ kg,
                                                  const __bf16* __restrict__ vtg,
                                                  __bf16* __restrict__ ao) {
  __shared__ __align__(64) __bf16 sP[8][16][32];
  const int lane = threadIdx.x & 31;
  const int wave = threadIdx.x >> 5;
  const int bh = blockIdx.x >> 3;   // b*H + head
  const int iblk = blockIdx.x & 7;
  const int i0 = iblk * 128 + wave * 16;
  const __bf16* Q = qg + bh * (NN * HD);
  const __bf16* K = kg + bh * (NN * HD);
  const __bf16* V = vtg + bh * (HD * NN);   // [64 x 1024] (d-major)
  const int r = lane & 15, h = lane >> 4;

  const v16bf qf0 = load_a_frag(Q, HD, i0, 0, lane);
  const v16bf qf1 = load_a_frag(Q, HD, i0, 32, lane);

  v8f O[4] = {};
  float Mrow[8], Lrow[8];
#pragma unroll
  for (int g = 0; g < 8; ++g) { Mrow[g] = -1e30f; Lrow[g] = 0.f; }

  for (int j0 = 0; j0 < NN; j0 += 32) {
    v8f s0 = {}, s1 = {};
    s0 = wmma_bf16(qf0, load_b_frag(K, HD, j0, 0, lane), s0);
    s0 = wmma_bf16(qf1, load_b_frag(K, HD, j0, 32, lane), s0);
    s1 = wmma_bf16(qf0, load_b_frag(K, HD, j0 + 16, 0, lane), s1);
    s1 = wmma_bf16(qf1, load_b_frag(K, HD, j0 + 16, 32, lane), s1);

#pragma unroll
    for (int g = 0; g < 8; ++g) {
      float mx = fmaxf(s0[g], s1[g]);
#pragma unroll
      for (int off = 8; off >= 1; off >>= 1)
        mx = fmaxf(mx, __shfl_xor(mx, off, 32));
      const float mnew = fmaxf(Mrow[g], mx);
      const float fg = __expf(Mrow[g] - mnew);
      const float p0 = __expf(s0[g] - mnew);
      const float p1 = __expf(s1[g] - mnew);
      float rs = p0 + p1;
#pragma unroll
      for (int off = 8; off >= 1; off >>= 1)
        rs += __shfl_xor(rs, off, 32);
      Lrow[g] = Lrow[g] * fg + rs;
      Mrow[g] = mnew;
#pragma unroll
      for (int ni = 0; ni < 4; ++ni) O[ni][g] *= fg;
      sP[wave][g + 8 * h][r] = (__bf16)p0;
      sP[wave][g + 8 * h][16 + r] = (__bf16)p1;
    }
    // DS ops are in-order per wave; keep the compiler from reordering and
    // make the store->load ordering explicit.
    asm volatile("s_wait_dscnt 0" ::: "memory");

    v16bf pf;
    {
      const __bf16* base = &sP[wave][r][h * 8];
      v8bf lo = *(const v8bf*)(base);
      v8bf hi = *(const v8bf*)(base + 16);
#pragma unroll
      for (int i = 0; i < 8; ++i) { pf[i] = lo[i]; pf[i + 8] = hi[i]; }
    }
#pragma unroll
    for (int ni = 0; ni < 4; ++ni)
      O[ni] = wmma_bf16(pf, load_b_frag(V, NN, ni * 16, j0, lane), O[ni]);
    asm volatile("s_wait_dscnt 0" ::: "memory");
  }

  const int b_ = bh / HH, head = bh % HH;
#pragma unroll
  for (int ni = 0; ni < 4; ++ni) {
#pragma unroll
    for (int g = 0; g < 8; ++g) {
      const int row = i0 + g + 8 * h;
      const int col = head * HD + ni * 16 + r;
      ao[(b_ * NN + row) * CC + col] = (__bf16)(O[ni][g] / Lrow[g]);
    }
  }
}

// ---------------------------------------------------------------------------
// Projection GEMM with bias: [8192 x 768] x [768 x 768] + b, f32 out.
// ---------------------------------------------------------------------------
__global__ __launch_bounds__(256) void proj_gemm(const __bf16* __restrict__ ao,
                                                 const __bf16* __restrict__ wp,
                                                 const float* __restrict__ bias,
                                                 float* __restrict__ out) {
  const int lane = threadIdx.x & 31;
  const int wave = threadIdx.x >> 5;
  const int wm = wave & 3, wn = wave >> 2;
  const int m0 = blockIdx.x * 128 + wm * 32;
  const int n0 = blockIdx.y * 128 + wn * 64;

  v8f acc[2][4] = {};
  for (int k0 = 0; k0 < CC; k0 += 32) {
    const v16bf a0 = load_a_frag(ao, CC, m0, k0, lane);
    const v16bf a1 = load_a_frag(ao, CC, m0 + 16, k0, lane);
#pragma unroll
    for (int ni = 0; ni < 4; ++ni) {
      const v16bf b = load_b_frag(wp, CC, n0 + ni * 16, k0, lane);
      acc[0][ni] = wmma_bf16(a0, b, acc[0][ni]);
      acc[1][ni] = wmma_bf16(a1, b, acc[1][ni]);
    }
  }

  const int r = lane & 15, h = lane >> 4;
#pragma unroll
  for (int mi = 0; mi < 2; ++mi) {
#pragma unroll
    for (int ni = 0; ni < 4; ++ni) {
#pragma unroll
      for (int g = 0; g < 8; ++g) {
        const int M = m0 + mi * 16 + g + 8 * h;
        const int Ncol = n0 + ni * 16 + r;
        out[(long)M * CC + Ncol] = acc[mi][ni][g] + bias[Ncol];
      }
    }
  }
}

// ---------------------------------------------------------------------------
// Launcher
// ---------------------------------------------------------------------------
extern "C" void kernel_launch(void* const* d_in, const int* in_sizes, int n_in,
                              void* d_out, int out_size, void* d_ws, size_t ws_size,
                              hipStream_t stream) {
  (void)in_sizes; (void)n_in; (void)out_size; (void)ws_size;
  const float* x      = (const float*)d_in[0];
  const float* w_qkv  = (const float*)d_in[1];
  const float* w_proj = (const float*)d_in[2];
  const float* b_proj = (const float*)d_in[3];
  float* out = (float*)d_out;

  char* p = (char*)d_ws;
  __bf16* xb     = (__bf16*)p; p += (size_t)TOKENS * CC * 2;
  __bf16* wqkvb  = (__bf16*)p; p += (size_t)C3 * CC * 2;
  __bf16* wprojb = (__bf16*)p; p += (size_t)CC * CC * 2;
  __bf16* qg     = (__bf16*)p; p += (size_t)BB * HH * NN * HD * 2;
  __bf16* kg     = (__bf16*)p; p += (size_t)BB * HH * NN * HD * 2;
  __bf16* vtg    = (__bf16*)p; p += (size_t)BB * HH * NN * HD * 2;
  __bf16* ao     = (__bf16*)p; p += (size_t)TOKENS * CC * 2;

  {
    int n4 = TOKENS * CC / 4;
    cvt_f32_bf16<<<(n4 + 255) / 256, 256, 0, stream>>>(x, xb, n4);
  }
  {
    int n4 = C3 * CC / 4;
    cvt_f32_bf16<<<(n4 + 255) / 256, 256, 0, stream>>>(w_qkv, wqkvb, n4);
  }
  {
    int n4 = CC * CC / 4;
    cvt_f32_bf16<<<(n4 + 255) / 256, 256, 0, stream>>>(w_proj, wprojb, n4);
  }
  qkv_gemm<<<dim3(TOKENS / 128, C3 / 128), 256, 0, stream>>>(xb, wqkvb, qg, kg, vtg);
  attn_fused<<<BB * HH * (NN / 128), 256, 0, stream>>>(qg, kg, vtg, ao);
  proj_gemm<<<dim3(TOKENS / 128, CC / 128), 256, 0, stream>>>(ao, wprojb, b_proj, out);
}
